// GATEncoder_88149908783550
// MI455X (gfx1250) — matrix operand order repaired and checked by reading, not compiled, not run on hardware
//
#include <hip/hip_runtime.h>
#include <hip/hip_bf16.h>

#define N_NODES 50000
#define N_EDGES 800000
#define ETOT    (N_EDGES + N_NODES)   // self-loops appended: edge e>=E is (e-E, e-E)
#define NC      128                    // H*C for every layer
#define KDIM    128
#define BSTRIDE 136                    // padded LDS row stride (bf16): 272B = 17*16B -> bank-conflict-free

typedef __attribute__((ext_vector_type(16))) __bf16 v16bf;
typedef __attribute__((ext_vector_type(8)))  __bf16 v8bf;
typedef __attribute__((ext_vector_type(8)))  float  v8f;

// ---------------- utility kernels ----------------

__global__ void k_f32_to_bf16(const float* __restrict__ in, __bf16* __restrict__ out, int n) {
    int i = blockIdx.x * blockDim.x + threadIdx.x;
    if (i < n) out[i] = (__bf16)in[i];
}

__global__ void k_fill_f32(float* __restrict__ p, float v, int n) {
    int i = blockIdx.x * blockDim.x + threadIdx.x;
    if (i < n) p[i] = v;
}

// ---- WMMA GEMM: C[M,128] = A[M,128] x B[128,128], bf16 in / f32 acc ----
// W is staged into padded LDS with CDNA5 async global->LDS copies (ASYNCcnt),
// B fragments are software-pipelined strided LDS column reads (8+8 bf16, row stride BSTRIDE) feeding v_wmma.

__global__ void __launch_bounds__(256)
k_gemm_bf16(const __bf16* __restrict__ A, const __bf16* __restrict__ Bg,
            float* __restrict__ C, int M) {
    __shared__ __bf16 Bs[KDIM * BSTRIDE];          // 34.8 KB of the WGP's 320 KB LDS

    // Async copy of W into padded LDS: 2048 x 16B segments across 256 lanes,
    // direct global->LDS (no VGPR bounce), tracked by ASYNCcnt.
    for (int i = threadIdx.x; i < (KDIM * NC) / 8; i += 256) {
        int row = i >> 4, seg = i & 15;
        unsigned      ldsa = (unsigned)(uintptr_t)(Bs + row * BSTRIDE + seg * 8);
        const __bf16* gsrc = Bg + row * NC + seg * 8;
        asm volatile("global_load_async_to_lds_b128 %0, %1, off"
                     :: "v"(ldsa), "v"(gsrc) : "memory");
    }
    asm volatile("s_wait_asynccnt 0x0" ::: "memory");
    __syncthreads();

    const int lane = threadIdx.x & 31;
    const int wave = threadIdx.x >> 5;
    const int tile = blockIdx.x * 8 + wave;        // 16-row tile index
    if (tile >= (M >> 4)) return;                  // uniform per wave; no barriers after this

    const int m   = (tile << 4) + (lane & 15);
    const int khi = (lane >> 4) & 1;               // K-half selector per ISA A-layout

    // Preload ALL four A fragments (K=0..127): 8 global b128 loads in one clause, one wait.
    v16bf afr[4];
#pragma unroll
    for (int kk = 0; kk < 4; ++kk) {
        const __bf16* ap = A + (size_t)m * KDIM + kk * 32 + 8 * khi;
        v8bf alo = *(const v8bf*)(ap);
        v8bf ahi = *(const v8bf*)(ap + 16);
#pragma unroll
        for (int i = 0; i < 8; ++i) { afr[kk][i] = alo[i]; afr[kk][i + 8] = ahi[i]; }
    }

    // 32 fragments: f = kk*8 + nt  ->  LDS rows kk*32 + {8h+i, 16+8h+i}, column nt*16 + (lane&15)
    const __bf16* bbase = Bs + (size_t)(8 * (lane >> 4)) * BSTRIDE + (lane & 15);
#define BADDR(f) (bbase + ((f) >> 3) * 32 * BSTRIDE + ((f) & 7) * 16)

    v8f acc[8] = {};                               // 8 column tiles of 16 -> full 128 cols
    v8bf blo, bhi;
#pragma unroll
    for (int i = 0; i < 8; ++i) { blo[i] = BADDR(0)[i * BSTRIDE]; bhi[i] = BADDR(0)[(16 + i) * BSTRIDE]; }
#pragma unroll
    for (int f = 0; f < 32; ++f) {
        v16bf bfrag;
#pragma unroll
        for (int i = 0; i < 8; ++i) { bfrag[i] = blo[i]; bfrag[i + 8] = bhi[i]; }
        if (f < 31) {                              // prefetch next fragment before the WMMA
#pragma unroll
            for (int i = 0; i < 8; ++i) { blo[i] = BADDR(f + 1)[i * BSTRIDE]; bhi[i] = BADDR(f + 1)[(16 + i) * BSTRIDE]; }
        }
        acc[f & 7] = __builtin_amdgcn_wmma_f32_16x16x32_bf16(
            false, afr[f >> 3], false, bfrag, (short)0, acc[f & 7], false, false);
    }
#undef BADDR

    // C/D layout: lane -> col (lane&15), VGPR v -> row v + 8*(lane>>4)
    const int crow = (tile << 4) + 8 * (lane >> 4);
    const int ccol = lane & 15;
#pragma unroll
    for (int nt = 0; nt < 8; ++nt)
#pragma unroll
        for (int v = 0; v < 8; ++v)
            C[(size_t)(crow + v) * NC + nt * 16 + ccol] = acc[nt][v];
}

// ---------------- attention kernels ----------------

__device__ __forceinline__ void atomicMaxF(float* addr, float val) {
    int* ai = (int*)addr;
    int old = __float_as_int(__builtin_nontemporal_load(addr));
    while (__int_as_float(old) < val) {
        int assumed = old;
        old = atomicCAS(ai, assumed, __float_as_int(val));
        if (old == assumed) break;
    }
}

// per (node, head): s = <h, a_src>, d = <h, a_dst>; init running max / denom
__global__ void k_attn_node(const float* __restrict__ h,
                            const float* __restrict__ a_src,
                            const float* __restrict__ a_dst,
                            float* __restrict__ s, float* __restrict__ d,
                            float* __restrict__ mx, float* __restrict__ dn,
                            int H, int C) {
    int i = blockIdx.x * blockDim.x + threadIdx.x;
    if (i >= N_NODES * H) return;
    int n  = i / H;
    int hh = i - n * H;
    const float* hp = h + (size_t)n * NC + hh * C;
    const float* as = a_src + hh * C;
    const float* ad = a_dst + hh * C;
    float ss = 0.f, dd = 0.f;
    for (int c = 0; c < C; ++c) { float v = hp[c]; ss += v * as[c]; dd += v * ad[c]; }
    s[i] = ss; d[i] = dd;
    mx[i] = -INFINITY; dn[i] = 0.0f;
}

// per (edge, head): e = leaky_relu(s[src]+d[dst]); segment max over dst
__global__ void k_edge_max(const int* __restrict__ ei,
                           const float* __restrict__ s, const float* __restrict__ d,
                           float* __restrict__ ebuf, float* __restrict__ mx, int H) {
    int idx = blockIdx.x * blockDim.x + threadIdx.x;
    if (idx >= ETOT * H) return;
    int e  = idx / H;
    int hh = idx - e * H;
    int sn = (e < N_EDGES) ? ei[e]           : (e - N_EDGES);
    int dv = (e < N_EDGES) ? ei[N_EDGES + e] : (e - N_EDGES);
    float v = s[sn * H + hh] + d[dv * H + hh];
    v = (v >= 0.f) ? v : 0.2f * v;               // leaky_relu, slope 0.2
    ebuf[idx] = v;
    atomicMaxF(&mx[dv * H + hh], v);
}

// per (edge, head): ex = exp(e - max[dst]); segment sum over dst
__global__ void k_edge_exp(const int* __restrict__ ei,
                           float* __restrict__ ebuf,
                           const float* __restrict__ mx, float* __restrict__ dn, int H) {
    int idx = blockIdx.x * blockDim.x + threadIdx.x;
    if (idx >= ETOT * H) return;
    int e  = idx / H;
    int hh = idx - e * H;
    int dv = (e < N_EDGES) ? ei[N_EDGES + e] : (e - N_EDGES);
    float ex = __expf(ebuf[idx] - mx[dv * H + hh]);
    ebuf[idx] = ex;
    atomicAdd(&dn[dv * H + hh], ex);
}

// per (edge, 4 channels): out[dst,c..c+3] += alpha * h[src,c..c+3]
// b128 gather of the source row; 4-channel group never crosses a head boundary.
__global__ void k_edge_aggr(const int* __restrict__ ei, const float* __restrict__ h,
                            const float* __restrict__ ebuf, const float* __restrict__ dn,
                            float* __restrict__ obuf, int H, int cshift) {
    int idx = blockIdx.x * blockDim.x + threadIdx.x;
    if (idx >= ETOT * (NC / 4)) return;
    int e  = idx >> 5;                           // edge
    int c0 = (idx & 31) << 2;                    // first of 4 channels
    int hh = c0 >> cshift;                       // head owning these channels
    int sn = (e < N_EDGES) ? ei[e]           : (e - N_EDGES);
    int dv = (e < N_EDGES) ? ei[N_EDGES + e] : (e - N_EDGES);
    float alpha = ebuf[e * H + hh] / (dn[dv * H + hh] + 1e-16f);
    const float4 hv = *(const float4*)(h + (size_t)sn * NC + c0);
    float* op = obuf + (size_t)dv * NC + c0;
    atomicAdd(op + 0, alpha * hv.x);
    atomicAdd(op + 1, alpha * hv.y);
    atomicAdd(op + 2, alpha * hv.z);
    atomicAdd(op + 3, alpha * hv.w);
}

// bias (+ReLU), emit bf16 for next layer's GEMM and/or fp32 final output
__global__ void k_finalize(const float* __restrict__ obuf, const float* __restrict__ bias,
                           float* __restrict__ of32, __bf16* __restrict__ obf, int relu) {
    int i = blockIdx.x * blockDim.x + threadIdx.x;
    if (i >= N_NODES * NC) return;
    float v = obuf[i] + bias[i & 127];
    if (relu) v = fmaxf(v, 0.f);
    if (of32) of32[i] = v;
    if (obf)  obf[i]  = (__bf16)v;
}

// ---------------- driver ----------------

extern "C" void kernel_launch(void* const* d_in, const int* in_sizes, int n_in,
                              void* d_out, int out_size, void* d_ws, size_t ws_size,
                              hipStream_t stream) {
    const float* x  = (const float*)d_in[0];
    const int*   ei = (const int*)d_in[1];
    const float* W[3]  = {(const float*)d_in[2], (const float*)d_in[6],  (const float*)d_in[10]};
    const float* As[3] = {(const float*)d_in[3], (const float*)d_in[7],  (const float*)d_in[11]};
    const float* Ad[3] = {(const float*)d_in[4], (const float*)d_in[8],  (const float*)d_in[12]};
    const float* Bi[3] = {(const float*)d_in[5], (const float*)d_in[9],  (const float*)d_in[13]};
    const int Hs[3] = {4, 4, 1};
    const int Cs[3] = {32, 32, 128};
    const int Sh[3] = {5, 5, 7};           // log2(C) -> channel->head shift

    // bump allocator over workspace (~81 MB total)
    char* p = (char*)d_ws;
    auto bump = [&](size_t bytes) -> void* {
        void* r = (void*)p; p += (bytes + 255) & ~(size_t)255; return r;
    };
    __bf16* xb   = (__bf16*)bump((size_t)N_NODES * NC * sizeof(__bf16));
    __bf16* Wb   = (__bf16*)bump((size_t)KDIM * NC * sizeof(__bf16));
    float*  hbuf = (float*) bump((size_t)N_NODES * NC * sizeof(float));
    float*  obuf = (float*) bump((size_t)N_NODES * NC * sizeof(float));
    float*  sb   = (float*) bump((size_t)N_NODES * 4 * sizeof(float));
    float*  db   = (float*) bump((size_t)N_NODES * 4 * sizeof(float));
    float*  mx   = (float*) bump((size_t)N_NODES * 4 * sizeof(float));
    float*  dn   = (float*) bump((size_t)N_NODES * 4 * sizeof(float));
    float*  exb  = (float*) bump((size_t)ETOT * 4 * sizeof(float));

    const int T = 256;
    k_f32_to_bf16<<<(N_NODES * NC + T - 1) / T, T, 0, stream>>>(x, xb, N_NODES * NC);

    for (int l = 0; l < 3; ++l) {
        const int H = Hs[l], C = Cs[l];
        k_f32_to_bf16<<<(KDIM * NC + T - 1) / T, T, 0, stream>>>(W[l], Wb, KDIM * NC);
        k_gemm_bf16<<<((N_NODES / 16) + 7) / 8, 256, 0, stream>>>(xb, Wb, hbuf, N_NODES);
        k_attn_node<<<(N_NODES * H + T - 1) / T, T, 0, stream>>>(hbuf, As[l], Ad[l],
                                                                 sb, db, mx, dn, H, C);
        k_edge_max<<<(ETOT * H + T - 1) / T, T, 0, stream>>>(ei, sb, db, exb, mx, H);
        k_edge_exp<<<(ETOT * H + T - 1) / T, T, 0, stream>>>(ei, exb, mx, dn, H);
        k_fill_f32<<<(N_NODES * NC + T - 1) / T, T, 0, stream>>>(obuf, 0.0f, N_NODES * NC);
        k_edge_aggr<<<(ETOT * (NC / 4) + T - 1) / T, T, 0, stream>>>(ei, hbuf, exb, dn,
                                                                    obuf, H, Sh[l]);
        k_finalize<<<(N_NODES * NC + T - 1) / T, T, 0, stream>>>(
            obuf, Bi[l],
            (l == 2) ? (float*)d_out : nullptr,
            (l < 2)  ? xb            : nullptr,
            (l < 2)  ? 1 : 0);
    }
}
